// GRUObservationCellLogvar_46901042872631
// MI455X (gfx1250) — compile-verified, matrix operand order checked by reading, not run on hardware
//
#include <hip/hip_runtime.h>
#include <math.h>

// Problem constants (from the reference)
#define BB 65536
#define HH 256
#define DD 32
#define PP 16
#define NN 32768
#define K1 512            // P*D
#define K2 256            // H
#define C3H 768           // 3H
#define TWO_LOG_C 1.8378770664093453f

typedef __bf16  bf16_t;
typedef __bf16  v8bf  __attribute__((ext_vector_type(8)));
typedef __bf16  v16bf __attribute__((ext_vector_type(16)));
typedef float   v8f   __attribute__((ext_vector_type(8)));

static __device__ __forceinline__ v16bf load_frag(const bf16_t* p0, const bf16_t* p1) {
    v8bf a = *(const v8bf*)p0;     // global_load_b128
    v8bf b = *(const v8bf*)p1;     // global_load_b128
    return __builtin_shufflevector(a, b, 0,1,2,3,4,5,6,7,8,9,10,11,12,13,14,15);
}

static __device__ __forceinline__ float sigmoidf_(float x) {
    return 1.0f / (1.0f + __expf(-x));
}

// ---------------------------------------------------------------------------
// Kernel 1: elementwise prep. One thread per (n,d).
//   - gaussian loss -> d_out losses region
//   - 4->16 per-feature matmul + relu + mask, written (bf16) into the
//     *scrambled* X matrix [N, 512] exactly matching the reference's
//     transpose(2,0,1)+reshape semantics:
//       (p, n, d) -> row n' = p*2048 + n/16, col = (n%16)*32 + d
// ---------------------------------------------------------------------------
__global__ __launch_bounds__(256) void prep_kernel(
    const float* __restrict__ p, const float* __restrict__ X_obs,
    const float* __restrict__ M_obs, const int* __restrict__ i_obs,
    const float* __restrict__ w_prep, const float* __restrict__ bias_prep,
    float* __restrict__ losses_out, bf16_t* __restrict__ Xmat)
{
    int tid = blockIdx.x * blockDim.x + threadIdx.x;     // N*D threads
    if (tid >= NN * DD) return;
    int n = tid >> 5;
    int d = tid & 31;
    int row = i_obs[n];

    float x      = X_obs[tid];
    float m      = M_obs[tid];
    float mean   = p[row * (2 * DD) + d];
    float logvar = p[row * (2 * DD) + DD + d];
    float sigma  = __expf(0.5f * logvar);
    float err    = (x - mean) / sigma;

    losses_out[tid] = 0.5f * ((err * err + logvar + TWO_LOG_C) * m);

    const float* w = w_prep + d * 4 * PP;    // [4][16]
    const float* b = bias_prep + d * PP;

    int nprime_lo = n >> 4;          // n/16
    int colbase   = (n & 15) * 32 + d;
#pragma unroll
    for (int pp = 0; pp < PP; ++pp) {
        float v = x * w[0 * PP + pp] + mean * w[1 * PP + pp]
                + logvar * w[2 * PP + pp] + err * w[3 * PP + pp] + b[pp];
        v = fmaxf(v, 0.0f) * m;
        int nprime = pp * (NN / 16) + nprime_lo;          // pp*2048 + n/16
        Xmat[nprime * K1 + colbase] = (bf16_t)v;
    }
}

// ---------------------------------------------------------------------------
// Kernel 2: gather h rows -> bf16 h_obs [N, 256]
// ---------------------------------------------------------------------------
__global__ __launch_bounds__(256) void gather_h_kernel(
    const float* __restrict__ h, const int* __restrict__ i_obs,
    bf16_t* __restrict__ Hobs)
{
    int tid = blockIdx.x * blockDim.x + threadIdx.x;     // N*H threads
    int n = tid >> 8;
    int c = tid & 255;
    Hobs[tid] = (bf16_t)h[i_obs[n] * HH + c];
}

// ---------------------------------------------------------------------------
// Kernel 3: pack weights into WMMA-B-friendly bf16 layout.
// For col-tile ct, k-step ks, lane l: 16 contiguous bf16 =
//   B[k][col] with col = ct*16 + (l&15),
//   k = ks*32 + (l>>4)*8 + (e<8 ? e : e+8)    (matches 16-bit operand layout)
// ---------------------------------------------------------------------------
__global__ __launch_bounds__(256) void pack_w_kernel(
    const float* __restrict__ gru_kernel, const float* __restrict__ gru_rec_kernel,
    bf16_t* __restrict__ Bx, bf16_t* __restrict__ Bh)
{
    int tid = blockIdx.x * blockDim.x + threadIdx.x;
    if (tid < 48 * 16 * 32 * 16) {           // Bx: K=512 -> 16 ksteps
        int e = tid & 15, lane = (tid >> 4) & 31, ks = (tid >> 9) & 15, ct = tid >> 13;
        int col = ct * 16 + (lane & 15);
        int k = ks * 32 + (lane >> 4) * 8 + (e < 8 ? e : e + 8);
        Bx[tid] = (bf16_t)gru_kernel[k * C3H + col];
    }
    if (tid < 48 * 8 * 32 * 16) {            // Bh: K=256 -> 8 ksteps
        int e = tid & 15, lane = (tid >> 4) & 31, ks = (tid >> 9) & 7, ct = tid >> 12;
        int col = ct * 16 + (lane & 15);
        int k = ks * 32 + (lane >> 4) * 8 + (e < 8 ? e : e + 8);
        Bh[tid] = (bf16_t)gru_rec_kernel[k * C3H + col];
    }
}

// ---------------------------------------------------------------------------
// Kernel 4: fused dual-GEMM (bf16 WMMA, f32 acc) + GRU gates + scatter.
// Block = 256 threads (8 waves), one 16-row stripe, all 768 columns.
// Wave w owns column tiles w*6 .. w*6+5 (48 tiles of 16 cols).
//   cols [0,512): single accumulator for mx+mh (z,r pre-activations)
//   cols [512,768): separate mx (xh) / mh (rh) accumulators
// ---------------------------------------------------------------------------
__global__ __launch_bounds__(256) void gemm_gru_kernel(
    const float* __restrict__ h, const int* __restrict__ i_obs,
    const bf16_t* __restrict__ Xmat, const bf16_t* __restrict__ Hobs,
    const bf16_t* __restrict__ Bx, const bf16_t* __restrict__ Bh,
    const float* __restrict__ bias_in, const float* __restrict__ bias_rec,
    float* __restrict__ h_out)
{
    __shared__ float S1[16][512];   // xz+rz , xr+rr  (cols 0..511)
    __shared__ float S2[16][256];   // xh             (cols 512..767)
    __shared__ float S3[16][256];   // rh             (cols 512..767)

    const int wg   = blockIdx.x;            // 0..2047 -> rows wg*16 .. +15
    const int wave = threadIdx.x >> 5;
    const int lane = threadIdx.x & 31;
    const int lrow = lane & 15;
    const int lhi  = lane >> 4;

    const bf16_t* Xrow = Xmat + (wg * 16 + lrow) * K1 + lhi * 8;
    const bf16_t* Hrow = Hobs + (wg * 16 + lrow) * K2 + lhi * 8;

    for (int t = 0; t < 6; ++t) {
        const int ct = wave * 6 + t;
        v8f accx = {};
        v8f acch = {};

        const bf16_t* bx = Bx + ((ct * 16) * 32 + lane) * 16;
        __builtin_prefetch(bx, 0, 1);
#pragma unroll
        for (int ks = 0; ks < 16; ++ks) {
            v16bf a = load_frag(Xrow + ks * 32, Xrow + ks * 32 + 16);
            v16bf b = load_frag(bx + ks * 512, bx + ks * 512 + 8);
            accx = __builtin_amdgcn_wmma_f32_16x16x32_bf16(
                false, a, false, b, (short)0, accx, false, false);
        }

        const bf16_t* bh = Bh + ((ct * 8) * 32 + lane) * 16;
#pragma unroll
        for (int ks = 0; ks < 8; ++ks) {
            v16bf a = load_frag(Hrow + ks * 32, Hrow + ks * 32 + 16);
            v16bf b = load_frag(bh + ks * 512, bh + ks * 512 + 8);
            acch = __builtin_amdgcn_wmma_f32_16x16x32_bf16(
                false, a, false, b, (short)0, acch, false, false);
        }

        const int col = ct * 16 + lrow;
#pragma unroll
        for (int v = 0; v < 8; ++v) {
            const int m = v + lhi * 8;      // C/D layout: VGPR v, lanegroup -> row
            if (ct < 32) {
                S1[m][col] = accx[v] + acch[v];
            } else {
                S2[m][col - 512] = accx[v];
                S3[m][col - 512] = acch[v];
            }
        }
    }
    __syncthreads();

    // GRU gates: thread -> (row m, 16 columns)
    const int m  = threadIdx.x >> 4;
    const int cb = (threadIdx.x & 15) * 16;
    const int ng = wg * 16 + m;
    const int dest = i_obs[ng];
    const float* hp = h + (size_t)dest * HH;
    float* ho = h_out + (size_t)dest * HH;
#pragma unroll 4
    for (int j = 0; j < 16; ++j) {
        const int c = cb + j;
        float z  = sigmoidf_(S1[m][c]       + bias_in[c]       + bias_rec[c]);
        float r  = sigmoidf_(S1[m][256 + c] + bias_in[256 + c] + bias_rec[256 + c]);
        float xh = S2[m][c] + bias_in[512 + c];
        float rh = S3[m][c] + bias_rec[512 + c];
        float hh = tanhf(xh + r * rh);
        float hv = hp[c];
        ho[c] = z * hv + (1.0f - z) * hh;
    }
}

// ---------------------------------------------------------------------------
extern "C" void kernel_launch(void* const* d_in, const int* in_sizes, int n_in,
                              void* d_out, int out_size, void* d_ws, size_t ws_size,
                              hipStream_t stream) {
    const float* h        = (const float*)d_in[0];
    const float* p        = (const float*)d_in[1];
    const float* X_obs    = (const float*)d_in[2];
    const float* M_obs    = (const float*)d_in[3];
    const int*   i_obs    = (const int*)d_in[4];
    const float* w_prep   = (const float*)d_in[5];
    const float* b_prep   = (const float*)d_in[6];
    const float* gru_k    = (const float*)d_in[7];
    const float* gru_rk   = (const float*)d_in[8];
    const float* gb_in    = (const float*)d_in[9];
    const float* gb_rec   = (const float*)d_in[10];

    float* h_out      = (float*)d_out;                       // [B, H]
    float* losses_out = (float*)d_out + (size_t)BB * HH;     // [N, D]

    char* ws = (char*)d_ws;
    bf16_t* Xmat = (bf16_t*)(ws);                                        // 32 MB
    bf16_t* Hobs = (bf16_t*)(ws + (size_t)NN * K1 * 2);                  // 16 MB
    bf16_t* Bx   = (bf16_t*)(ws + (size_t)NN * K1 * 2 + (size_t)NN * K2 * 2);
    bf16_t* Bh   = (bf16_t*)((char*)Bx + (size_t)48 * 16 * 32 * 16 * 2);

    // h_out starts as a copy of h; GRU rows are scattered over it afterwards.
    hipMemcpyAsync(h_out, h, (size_t)BB * HH * sizeof(float),
                   hipMemcpyDeviceToDevice, stream);

    prep_kernel<<<(NN * DD) / 256, 256, 0, stream>>>(
        p, X_obs, M_obs, i_obs, w_prep, b_prep, losses_out, Xmat);

    gather_h_kernel<<<(NN * HH) / 256, 256, 0, stream>>>(h, i_obs, Hobs);

    pack_w_kernel<<<(48 * 16 * 32 * 16) / 256, 256, 0, stream>>>(
        gru_k, gru_rk, Bx, Bh);

    gemm_gru_kernel<<<NN / 16, 256, 0, stream>>>(
        h, i_obs, Xmat, Hobs, Bx, Bh, gb_in, gb_rec, h_out);
}
